// MSDeformAttn_74629351735672
// MI455X (gfx1250) — compile-verified, matrix operand order checked by reading, not run on hardware
//
#include <hip/hip_runtime.h>
#include <hip/hip_bf16.h>

// MI455X (gfx1250) MS-deformable attention.
// Memory-bound (~12.5 GFLOP vs ~180 MB traffic @ 23.3 TB/s HBM, mostly
// L2-resident in 192 MB). GEMMs use v_wmma_f32_16x16x32_f16 (f16 in, f32 acc).
// Weights are pre-transposed + pre-converted to f16 once (311 KB total) so B
// fragments are two contiguous b128 loads per K-step (no cvt, no addr chains).
// Each wave computes a 32x32 slab (2 M-tiles x 2 N-tiles, 4 WMMAs/K-step);
// K is a compile-time constant so the K-loop fully unrolls and the scheduler
// can pipeline loads across WMMAs instead of wait-converting each b128.

typedef __attribute__((ext_vector_type(16))) _Float16 v16h;
typedef __attribute__((ext_vector_type(8)))  _Float16 v8h;
typedef __attribute__((ext_vector_type(8)))  float    v8f;

#define B_   4
#define LQ_  10000
#define D_   256
#define H_   8
#define P_   4
#define CD_  32
#define HW_  100
#define LIN_ (HW_ * HW_)
#define MROWS_ (B_ * LQ_)   // 40000 == B_*LIN_; divisible by 64

// ---------------------------------------------------------------------------
// Weight prep: Wt[n][k] (f16, row-major N x K) = (f16)W[k][n] (f32, K x N).
// Writes are fully coalesced; reads strided but the matrices are tiny.
// ---------------------------------------------------------------------------
__global__ __launch_bounds__(256)
void transpose_f16(const float* __restrict__ W, _Float16* __restrict__ Wt,
                   int K, int N)
{
    const int idx = blockIdx.x * blockDim.x + threadIdx.x;
    if (idx >= K * N) return;
    const int n = idx / K;
    const int k = idx - n * K;
    Wt[idx] = (_Float16)W[(size_t)k * N + n];
}

// ---------------------------------------------------------------------------
// C[M,N] = A[M,KC] @ W[KC,N] + bias[N]; MASKED: zero rows where mask[row]!=0.
// Wt is the f16-transposed weight (N x KC). One wave -> 32x32 output slab.
// A fragment (16-bit A 16x32, ISA 7.12.2): lane L -> row (L&15),
//   K-base kb = (L&16)?8:0; fragment halves 0..7 = K kb+k0+0..7,
//   halves 8..15 = K kb+k0+16..23 (contiguous in memory -> b128 loads).
// B fragment: mirror mapping with lane -> column; identical K pattern, so a
//   row of Wt gives the fragment as two contiguous v8h (16B) loads.
// C/D: VGPR r -> row r (lanes 0-15) / row r+8 (lanes 16-31), col = lane&15.
// ---------------------------------------------------------------------------
template <bool MASKED, int KC>
__global__ __launch_bounds__(256)
void gemm_bias_wmma(const float* __restrict__ A, const _Float16* __restrict__ Wt,
                    const float* __restrict__ bias, float* __restrict__ C,
                    int M, int N, const unsigned char* __restrict__ mask)
{
    const int lane = threadIdx.x & 31;
    const int wave = threadIdx.x >> 5;
    const int wpb  = blockDim.x >> 5;
    const long group = (long)blockIdx.x * wpb + wave;   // 32-row x 32-col slab
    const int  ntn   = N >> 5;                          // 32-wide column groups
    const long ngroups = (long)(M >> 5) * ntn;
    if (group >= ngroups) return;                // wave-uniform: EXEC stays all-1s

    const int m0 = (int)(group / ntn) << 5;
    const int n0 = (int)(group % ntn) << 5;

    const int rl = lane & 15;
    const int kb = (lane & 16) ? 8 : 0;          // per-half-wave K base

    const float* a0 = A + (size_t)(m0 + rl) * KC + kb;
    const float* a1 = a0 + (size_t)16 * KC;                      // m-tile 1

    const _Float16* w0 = Wt + (size_t)(n0 + rl) * KC + kb;       // n-tile 0
    const _Float16* w1 = w0 + (size_t)16 * KC;                   // n-tile 1

    v8f acc[2][2] = {{{}, {}}, {{}, {}}};

#pragma unroll
    for (int k0 = 0; k0 < KC; k0 += 32) {
        // B fragments: two contiguous 16B loads each, no conversion needed
        const v8h b0lo = *(const v8h*)(w0 + k0);
        const v8h b0hi = *(const v8h*)(w0 + k0 + 16);
        const v8h b1lo = *(const v8h*)(w1 + k0);
        const v8h b1hi = *(const v8h*)(w1 + k0 + 16);
        v16h bf[2];
        bf[0] = __builtin_shufflevector(b0lo, b0hi,
                    0,1,2,3,4,5,6,7,8,9,10,11,12,13,14,15);
        bf[1] = __builtin_shufflevector(b1lo, b1hi,
                    0,1,2,3,4,5,6,7,8,9,10,11,12,13,14,15);

        // A fragments: contiguous K-pairs -> b128-mergeable 8B vector loads
        v16h af[2];
        const float* ap[2] = {a0 + k0, a1 + k0};
#pragma unroll
        for (int t = 0; t < 2; ++t) {
#pragma unroll
            for (int j = 0; j < 8; ++j) {
                const int k = (j < 4) ? (2 * j) : (16 + 2 * (j - 4));
                const float2 v = *(const float2*)(ap[t] + k);
                af[t][2 * j]     = (_Float16)v.x;
                af[t][2 * j + 1] = (_Float16)v.y;
            }
        }
#pragma unroll
        for (int t = 0; t < 2; ++t)
#pragma unroll
            for (int u = 0; u < 2; ++u)
                acc[t][u] = __builtin_amdgcn_wmma_f32_16x16x32_f16(
                    /*neg_a=*/false, af[t], /*neg_b=*/false, bf[u],
                    /*c_mod=*/(short)0, acc[t][u],
                    /*reuse_a=*/false, /*reuse_b=*/false);
    }

    const float bn0 = bias[n0 + rl];
    const float bn1 = bias[n0 + 16 + rl];
#pragma unroll
    for (int t = 0; t < 2; ++t) {
        const int rb = m0 + 16 * t + ((lane & 16) ? 8 : 0);
#pragma unroll
        for (int i = 0; i < 8; ++i) {
            const int row = rb + i;
            float v0 = acc[t][0][i] + bn0;
            float v1 = acc[t][1][i] + bn1;
            if (MASKED) {
                const bool z = (mask[row] != 0);   // branch-free select
                v0 = z ? 0.0f : v0;
                v1 = z ? 0.0f : v1;
            }
            C[(size_t)row * N + n0 + rl]      = v0;
            C[(size_t)row * N + n0 + 16 + rl] = v1;
        }
    }
}

// ---------------------------------------------------------------------------
// Sampling stage: one wave per (b, q, head); lane = channel within head
// (Cd = 32 == wave32 width -> every corner gather is a coalesced 128B load).
// Softmax over the 4 points and all coordinates are wave-uniform scalar work.
// ---------------------------------------------------------------------------
__global__ __launch_bounds__(256)
void msda_sample(const float* __restrict__ value,  // (B, LIN, D)
                 const float* __restrict__ off,    // (B*LQ, 64)  [h][p][2]
                 const float* __restrict__ alog,   // (B*LQ, 32)  [h][p]
                 const float* __restrict__ refp,   // (B*LQ, 4)   cx,cy,w,h
                 float* __restrict__ mid)          // (B, LQ, D)
{
    const int lane = threadIdx.x & 31;
    const int wid  = blockIdx.x * (blockDim.x >> 5) + (threadIdx.x >> 5);
    if (wid >= B_ * LQ_ * H_) return;
    const int h  = wid & 7;
    const int bq = wid >> 3;
    const int b  = bq / LQ_;

    const float* rp = refp + (size_t)bq * 4;
    const float cx = rp[0], cy = rp[1];
    const float sx = rp[2] * 0.125f;            // (1/P) * 0.5 = 0.125
    const float sy = rp[3] * 0.125f;

    // softmax over the L*P = 4 attention logits of this head
    const float* al = alog + (size_t)bq * 32 + h * 4;
    const float l0 = al[0], l1 = al[1], l2 = al[2], l3 = al[3];
    const float mx = fmaxf(fmaxf(l0, l1), fmaxf(l2, l3));
    const float e0 = __expf(l0 - mx), e1 = __expf(l1 - mx);
    const float e2 = __expf(l2 - mx), e3 = __expf(l3 - mx);
    const float rs = 1.0f / (e0 + e1 + e2 + e3);
    const float awv[4] = {e0 * rs, e1 * rs, e2 * rs, e3 * rs};

    const float* of = off + (size_t)bq * 64 + h * 8;
    const float* vb = value + (size_t)b * LIN_ * D_ + h * CD_ + lane;

    float acc = 0.0f;
#pragma unroll
    for (int p = 0; p < P_; ++p) {
        const float x = (cx + of[2 * p]     * sx) * (float)HW_ - 0.5f;
        const float y = (cy + of[2 * p + 1] * sy) * (float)HW_ - 0.5f;
        const float xf = floorf(x), yf = floorf(y);
        const int x0 = (int)xf, y0 = (int)yf, x1 = x0 + 1, y1 = y0 + 1;
        const float wx = x - xf, wy = y - yf;
        const float w00 = (1.0f - wx) * (1.0f - wy);
        const float w01 = (1.0f - wx) * wy;
        const float w10 = wx * (1.0f - wy);
        const float w11 = wx * wy;

        float s = 0.0f;   // invalid corners contribute 0 (matches ref valid())
        if (x0 >= 0 && x0 < HW_ && y0 >= 0 && y0 < HW_)
            s += w00 * vb[(size_t)(y0 * HW_ + x0) * D_];
        if (x0 >= 0 && x0 < HW_ && y1 >= 0 && y1 < HW_)
            s += w01 * vb[(size_t)(y1 * HW_ + x0) * D_];
        if (x1 >= 0 && x1 < HW_ && y0 >= 0 && y0 < HW_)
            s += w10 * vb[(size_t)(y0 * HW_ + x1) * D_];
        if (x1 >= 0 && x1 < HW_ && y1 >= 0 && y1 < HW_)
            s += w11 * vb[(size_t)(y1 * HW_ + x1) * D_];
        acc += awv[p] * s;
    }

    mid[(size_t)bq * D_ + h * CD_ + lane] = acc;
}

// ---------------------------------------------------------------------------
extern "C" void kernel_launch(void* const* d_in, const int* in_sizes, int n_in,
                              void* d_out, int out_size, void* d_ws, size_t ws_size,
                              hipStream_t stream)
{
    (void)in_sizes; (void)n_in; (void)out_size; (void)ws_size;

    const float*          query = (const float*)d_in[0];
    const float*          refp  = (const float*)d_in[1];
    const float*          inp   = (const float*)d_in[2];
    // d_in[3], d_in[4] (spatial shapes / level start) are redundant single-level.
    const unsigned char*  pmask = (const unsigned char*)d_in[5];
    const float* Wv   = (const float*)d_in[6];
    const float* bv   = (const float*)d_in[7];
    const float* Woff = (const float*)d_in[8];
    const float* boff = (const float*)d_in[9];
    const float* Wa   = (const float*)d_in[10];
    const float* ba   = (const float*)d_in[11];
    const float* Wout = (const float*)d_in[12];
    const float* bout = (const float*)d_in[13];
    float* out = (float*)d_out;

    // workspace: value | off | alog | mid (f32), then f16 transposed weights.
    // f32 region = 40000*(256+64+32+256)*4 = 97,280,000 B (16B aligned).
    char* ws = (char*)d_ws;
    float* ws_value = (float*)(ws);
    float* ws_off   = (float*)(ws + (size_t)MROWS_ * (D_)           * 4);
    float* ws_alog  = (float*)(ws + (size_t)MROWS_ * (D_ + 64)      * 4);
    float* ws_mid   = (float*)(ws + (size_t)MROWS_ * (D_ + 64 + 32) * 4);
    char*  wbase    = ws + (size_t)MROWS_ * (D_ + 64 + 32 + D_) * 4;
    _Float16* WvT   = (_Float16*)(wbase);                          // 256x256
    _Float16* WoffT = (_Float16*)(wbase + 2 * (256 * 256));        // 64x256
    _Float16* WaT   = (_Float16*)(wbase + 2 * (256 * 256 + 64 * 256));   // 32x256
    _Float16* WoutT = (_Float16*)(wbase + 2 * (256 * 256 + 64 * 256 + 32 * 256));

    const int WPB = 8;                       // 256-thread blocks = 8 waves
    auto blocks = [](long work, int per) { return (int)((work + per - 1) / per); };

    // 0) pre-transpose + f16-convert weights (tiny: 311 KB total)
    transpose_f16<<<blocks(256 * 256, 256), 256, 0, stream>>>(Wv,   WvT,   256, 256);
    transpose_f16<<<blocks(256 * 64,  256), 256, 0, stream>>>(Woff, WoffT, 256, 64);
    transpose_f16<<<blocks(256 * 32,  256), 256, 0, stream>>>(Wa,   WaT,   256, 32);
    transpose_f16<<<blocks(256 * 256, 256), 256, 0, stream>>>(Wout, WoutT, 256, 256);

    // 1) value projection (masked): (40000x256) @ (256x256)
    {
        long g = (long)(MROWS_ / 32) * (D_ / 32);
        gemm_bias_wmma<true, 256><<<blocks(g, WPB), 256, 0, stream>>>(
            inp, WvT, bv, ws_value, MROWS_, D_, pmask);
    }
    // 2) sampling-offset projection: (40000x256) @ (256x64)
    {
        long g = (long)(MROWS_ / 32) * (64 / 32);
        gemm_bias_wmma<false, 256><<<blocks(g, WPB), 256, 0, stream>>>(
            query, WoffT, boff, ws_off, MROWS_, 64, nullptr);
    }
    // 3) attention-logit projection: (40000x256) @ (256x32)
    {
        long g = (long)(MROWS_ / 32) * (32 / 32);
        gemm_bias_wmma<false, 256><<<blocks(g, WPB), 256, 0, stream>>>(
            query, WaT, ba, ws_alog, MROWS_, 32, nullptr);
    }
    // 4) bilinear sample + softmax-weighted reduction
    {
        long waves = (long)B_ * LQ_ * H_;    // 320000 waves, one per (b,q,h)
        msda_sample<<<blocks(waves, WPB), 256, 0, stream>>>(
            ws_value, ws_off, ws_alog, refp, ws_mid);
    }
    // 5) output projection: (40000x256) @ (256x256)
    {
        long g = (long)(MROWS_ / 32) * (D_ / 32);
        gemm_bias_wmma<false, 256><<<blocks(g, WPB), 256, 0, stream>>>(
            ws_mid, WoutT, bout, out, MROWS_, D_, nullptr);
    }
}
